// MesoNet_11742440588099
// MI455X (gfx1250) — matrix operand order repaired and checked by reading, not compiled, not run on hardware
//
#include <hip/hip_runtime.h>

#define NN   16384      // nodes
#define NE   32768      // edges
#define DD   128        // feature dim
#define EDIM 10         // edge attr dim
#define EHB  33         // 32 hidden + 1 bias channel
#define HSTR 34         // padded row stride for h
#define KTOT (EHB * DD) // 4224 : K of the big GEMM

typedef _Float16 v16h __attribute__((ext_vector_type(16)));
typedef _Float16 v8h  __attribute__((ext_vector_type(8)));
typedef float    v8f  __attribute__((ext_vector_type(8)));
typedef int      v4i  __attribute__((ext_vector_type(4)));

#define AS1 __attribute__((address_space(1)))
#define AS3 __attribute__((address_space(3)))

#if defined(__gfx1250__) && __has_builtin(__builtin_amdgcn_global_load_async_to_lds_b128)
#define HAVE_ASYNC_LDS 1
#else
#define HAVE_ASYNC_LDS 0
#endif

static __device__ __forceinline__ v8h splat8(_Float16 v) {
    v8h r = {v, v, v, v, v, v, v, v};
    return r;
}
static __device__ __forceinline__ v16h cat16(v8h a, v8h b) {
    return __builtin_shufflevector(a, b, 0, 1, 2, 3, 4, 5, 6, 7,
                                   8, 9, 10, 11, 12, 13, 14, 15);
}
static __device__ __forceinline__ v8f wmma16(v16h a, v16h b, v8f c) {
    // v_wmma_f32_16x16x32_f16  D = A*B + C
    return __builtin_amdgcn_wmma_f32_16x16x32_f16(false, a, false, b,
                                                  (short)0, c, false, false);
}

// stage nchunks x 16B from global into LDS (256 threads), async if available
template <int NCHUNK>
static __device__ __forceinline__ void stage_tile(const _Float16* __restrict__ g,
                                                  _Float16* __restrict__ s, int tid) {
#if HAVE_ASYNC_LDS
#pragma unroll
    for (int i = 0; i < NCHUNK / 256; ++i) {
        const int c = (tid + 256 * i) * 8;
        __builtin_amdgcn_global_load_async_to_lds_b128(
            (AS1 v4i*)(g + c), (AS3 v4i*)(s + c), 0, 0);
    }
    asm volatile("s_wait_asynccnt 0" ::: "memory");
#else
    const v8h* gv = (const v8h*)g;
    v8h* sv = (v8h*)s;
#pragma unroll
    for (int i = 0; i < NCHUNK / 256; ++i) sv[tid + 256 * i] = gv[tid + 256 * i];
#endif
}

// ---------------------------------------------------------------- helpers ---
__global__ void k_zero(float* __restrict__ p, int n) {
    int i = blockIdx.x * 256 + threadIdx.x;
    if (i < n) p[i] = 0.0f;
}

__global__ void k_count(const int* __restrict__ dste, float* __restrict__ cnt) {
    int e = blockIdx.x * 256 + threadIdx.x;
    if (e < NE) atomicAdd(cnt + dste[e], 1.0f);
}

// h[e,k] = relu(edge_attr[e] . l1w[:,k] + l1b[k]),  h[e,32] = 1 (l2-bias channel)
__global__ void k_h(const float* __restrict__ ea, const float* __restrict__ w,
                    const float* __restrict__ b, _Float16* __restrict__ h) {
    int id = blockIdx.x * 256 + threadIdx.x;
    if (id >= NE * EHB) return;
    int e = id / EHB, k = id - e * EHB;
    float v;
    if (k == 32) {
        v = 1.0f;
    } else {
        v = b[k];
        const float* a = ea + (size_t)e * EDIM;
#pragma unroll
        for (int j = 0; j < EDIM; ++j) v += a[j] * w[j * 32 + k];
        v = fmaxf(v, 0.0f);
    }
    h[(size_t)e * HSTR + k] = (_Float16)v;
}

// T^T[o][k*128+i] = l2w[k, i*128+o] (k<32) else l2b[i*128+o]   (f16)
__global__ void k_tt(const float* __restrict__ l2w, const float* __restrict__ bl2,
                     _Float16* __restrict__ TT) {
    int id = blockIdx.x * 256 + threadIdx.x;
    if (id >= DD * KTOT) return;
    int o = id / KTOT, ki = id - o * KTOT;
    int k = ki >> 7, i = ki & 127;
    float v = (k < 32) ? l2w[(size_t)k * (DD * DD) + i * DD + o]
                       : bl2[(size_t)i * DD + o];
    TT[id] = (_Float16)v;
}

// root^T[o][i] = root[i][o] (f16)
__global__ void k_rt(const float* __restrict__ rw, _Float16* __restrict__ rT) {
    int id = blockIdx.x * 256 + threadIdx.x;
    if (id >= DD * DD) return;
    int o = id >> 7, i = id & 127;
    rT[id] = (_Float16)rw[(size_t)i * DD + o];
}

__global__ void k_cvt(const float* __restrict__ x, _Float16* __restrict__ xf) {
    int id = blockIdx.x * 256 + threadIdx.x;
    if (id >= NN * DD) return;
    xf[id] = (_Float16)x[id];
}

// xs[e] = xf16[src[e]]  (v8h granularity)
__global__ void k_gather(const _Float16* __restrict__ xf, const int* __restrict__ srce,
                         _Float16* __restrict__ xs) {
    int id = blockIdx.x * 256 + threadIdx.x;
    if (id >= NE * 16) return;
    int e = id >> 4, c = id & 15;
    int s = srce[e];
    *(v8h*)(xs + (size_t)e * DD + c * 8) =
        *(const v8h*)(xf + (size_t)s * DD + c * 8);
}

// ------------------------- big GEMM: msg[e,:] = (h_e (x) x_e) @ T, scatter ---
// block = 256 thr = 8 waves; block covers 128 edges (8 etiles).
// wave w: etiles (w>>2)*4 .. +3, N-tiles (w&3)*2 .. +1  -> 8 accumulators.
// Per K-chunk: 4 A-fragments (32 pk_mul) + 2 B-fragments feed 8 WMMAs
// -> 4 VALU pk_muls per WMMA (vs 8 in the naive mapping).
__global__ __launch_bounds__(256) void k_gemm_msg(
    const _Float16* __restrict__ xs, const _Float16* __restrict__ hb,
    const _Float16* __restrict__ TT, const int* __restrict__ dste,
    float* __restrict__ agg) {
    __shared__ alignas(16) _Float16 sx[128 * DD];   // 32 KB gathered x tile
    __shared__ _Float16 sh[128 * HSTR];             // 8.5 KB h tile
    __shared__ int sdst[128];

    const int tid = threadIdx.x;
    const int eb  = blockIdx.x * 128;

    stage_tile<2048>(xs + (size_t)eb * DD, sx, tid);
    for (int i = tid; i < 128 * HSTR; i += 256) sh[i] = hb[(size_t)eb * HSTR + i];
    if (tid < 128) sdst[tid] = dste[eb + tid];
    __syncthreads();

    const int lane = tid & 31;
    const int wv   = tid >> 5;
    const int r    = lane & 15;
    const int half = lane >> 4;
    const int kb   = half * 16;    // A-fragment K split: {0-15}|{16-31}
    const int etb  = (wv >> 2) * 4; // first etile of this wave
    const int ng   = wv & 3;        // N-tile pair

    v8f ac[4][2];
#pragma unroll
    for (int t = 0; t < 4; ++t)
#pragma unroll
        for (int u = 0; u < 2; ++u) ac[t][u] = (v8f){0, 0, 0, 0, 0, 0, 0, 0};

    const _Float16* trow0 = TT + (size_t)((ng * 2 + 0) * 16 + r) * KTOT;
    const _Float16* trow1 = TT + (size_t)((ng * 2 + 1) * 16 + r) * KTOT;

    for (int kk = 0; kk < EHB; ++kk) {          // 33 h-channels
        if (kk + 1 < EHB) {
            __builtin_prefetch(trow0 + (kk + 1) * DD);
            __builtin_prefetch(trow1 + (kk + 1) * DD);
        }
        v8h hs[4];
#pragma unroll
        for (int t = 0; t < 4; ++t)
            hs[t] = splat8(sh[((etb + t) * 16 + r) * HSTR + kk]);
#pragma unroll
        for (int q = 0; q < 4; ++q) {           // 4 K-chunks of 32 per channel
            const int c = kk * 4 + q;
            // B fragments: per half-lane group 16 contiguous K values
            const _Float16* bp0 = trow0 + c * 32 + half * 16;
            const _Float16* bp1 = trow1 + c * 32 + half * 16;
            v16h bf0 = cat16(*(const v8h*)bp0, *(const v8h*)(bp0 + 8));
            v16h bf1 = cat16(*(const v8h*)bp1, *(const v8h*)(bp1 + 8));
            const int i0 = q * 32 + kb;
            v16h af[4];
#pragma unroll
            for (int t = 0; t < 4; ++t) {
                const _Float16* ap = sx + ((etb + t) * 16 + r) * DD + i0;
                af[t] = cat16((*(const v8h*)ap) * hs[t],
                              (*(const v8h*)(ap + 8)) * hs[t]);
            }
#pragma unroll
            for (int t = 0; t < 4; ++t) {
                ac[t][0] = wmma16(af[t], bf0, ac[t][0]);
                ac[t][1] = wmma16(af[t], bf1, ac[t][1]);
            }
        }
    }

    // scatter-add straight from accumulators (C layout: M=j+8*half, N=lane&15)
#pragma unroll
    for (int t = 0; t < 4; ++t)
#pragma unroll
        for (int u = 0; u < 2; ++u) {
            const int n = (ng * 2 + u) * 16 + r;
#pragma unroll
            for (int j = 0; j < 8; ++j) {
                const int m = j + 8 * half;
                atomicAdd(agg + (size_t)sdst[(etb + t) * 16 + m] * DD + n,
                          ac[t][u][j]);
            }
        }
}

// -------- root GEMM + scatter-mean combine: x' = relu(x@root + agg/cnt + b) --
__global__ __launch_bounds__(256) void k_root_combine(
    _Float16* __restrict__ xf, const _Float16* __restrict__ rT,
    const float* __restrict__ agg, const float* __restrict__ cnt,
    const float* __restrict__ bias, float* __restrict__ out,
    const int do_relu, const int wr_out) {
    __shared__ alignas(16) _Float16 sx[64 * DD];

    const int tid = threadIdx.x;
    const int rb  = blockIdx.x * 64;
    stage_tile<1024>(xf + (size_t)rb * DD, sx, tid);
    __syncthreads();   // block's xf rows now in LDS -> safe to overwrite later

    const int lane = tid & 31;
    const int wv   = tid >> 5;
    const int r    = lane & 15;
    const int half = lane >> 4;
    const int kb   = half * 16;

    v8f ac[4];
#pragma unroll
    for (int t = 0; t < 4; ++t) ac[t] = (v8f){0, 0, 0, 0, 0, 0, 0, 0};

    const _Float16* brow = rT + (size_t)(wv * 16 + r) * DD;
#pragma unroll
    for (int c = 0; c < 4; ++c) {
        const _Float16* bp = brow + c * 32 + half * 16;
        v16h bf = cat16(*(const v8h*)bp, *(const v8h*)(bp + 8));
        const int i0 = c * 32 + kb;
        v16h af[4];
#pragma unroll
        for (int t = 0; t < 4; ++t) {
            const _Float16* ap = sx + (t * 16 + r) * DD + i0;
            af[t] = cat16(*(const v8h*)ap, *(const v8h*)(ap + 8));
        }
#pragma unroll
        for (int t = 0; t < 4; ++t) ac[t] = wmma16(af[t], bf, ac[t]);
    }

    const int n  = wv * 16 + r;
    const float bn = bias[n];
#pragma unroll
    for (int t = 0; t < 4; ++t)
#pragma unroll
        for (int j = 0; j < 8; ++j) {
            const int m   = j + 8 * half;
            const int row = rb + t * 16 + m;
            float v = ac[t][j] + agg[(size_t)row * DD + n] / fmaxf(cnt[row], 1.0f) + bn;
            if (do_relu) v = fmaxf(v, 0.0f);
            xf[(size_t)row * DD + n] = (_Float16)v;   // next layer input
            if (wr_out) out[(size_t)row * DD + n] = v;
        }
}

// -------------------------------------------------------------- launcher ----
extern "C" void kernel_launch(void* const* d_in, const int* in_sizes, int n_in,
                              void* d_out, int out_size, void* d_ws, size_t ws_size,
                              hipStream_t stream) {
    (void)in_sizes; (void)n_in; (void)out_size; (void)ws_size;

    const float* x    = (const float*)d_in[0];
    const int*   ei   = (const int*)d_in[1];
    const float* ea   = (const float*)d_in[2];
    const float* w1l1 = (const float*)d_in[3];
    const float* b1l1 = (const float*)d_in[4];
    const float* w1l2 = (const float*)d_in[5];
    const float* b1l2 = (const float*)d_in[6];
    const float* w1r  = (const float*)d_in[7];
    const float* b1   = (const float*)d_in[8];
    const float* w2l1 = (const float*)d_in[9];
    const float* b2l1 = (const float*)d_in[10];
    const float* w2l2 = (const float*)d_in[11];
    const float* b2l2 = (const float*)d_in[12];
    const float* w2r  = (const float*)d_in[13];
    const float* b2   = (const float*)d_in[14];
    float* out = (float*)d_out;

    const int* srce = ei;        // edge_index[0]
    const int* dste = ei + NE;   // edge_index[1]

    char* ws = (char*)d_ws;
    size_t off = 0;
    auto alloc = [&](size_t bytes) -> void* {
        void* p = ws + off;
        off += (bytes + 255) & ~(size_t)255;
        return p;
    };
    _Float16* xf = (_Float16*)alloc((size_t)NN * DD * 2);    // current x (f16)
    _Float16* xs = (_Float16*)alloc((size_t)NE * DD * 2);    // gathered x[src]
    _Float16* h1 = (_Float16*)alloc((size_t)NE * HSTR * 2);
    _Float16* h2 = (_Float16*)alloc((size_t)NE * HSTR * 2);
    _Float16* T1 = (_Float16*)alloc((size_t)DD * KTOT * 2);  // T^T for set 1
    _Float16* T2 = (_Float16*)alloc((size_t)DD * KTOT * 2);
    _Float16* r1 = (_Float16*)alloc((size_t)DD * DD * 2);    // root^T
    _Float16* r2 = (_Float16*)alloc((size_t)DD * DD * 2);
    float* agg = (float*)alloc((size_t)NN * DD * 4);
    float* cnt = (float*)alloc((size_t)NN * 4);

    dim3 B(256);
    k_zero<<<(NN + 255) / 256, B, 0, stream>>>(cnt, NN);
    k_count<<<(NE + 255) / 256, B, 0, stream>>>(dste, cnt);
    k_h<<<(NE * EHB + 255) / 256, B, 0, stream>>>(ea, w1l1, b1l1, h1);
    k_h<<<(NE * EHB + 255) / 256, B, 0, stream>>>(ea, w2l1, b2l1, h2);
    k_tt<<<(DD * KTOT + 255) / 256, B, 0, stream>>>(w1l2, b1l2, T1);
    k_tt<<<(DD * KTOT + 255) / 256, B, 0, stream>>>(w2l2, b2l2, T2);
    k_rt<<<(DD * DD + 255) / 256, B, 0, stream>>>(w1r, r1);
    k_rt<<<(DD * DD + 255) / 256, B, 0, stream>>>(w2r, r2);
    k_cvt<<<(NN * DD + 255) / 256, B, 0, stream>>>(x, xf);

    for (int l = 0; l < 3; ++l) {
        const _Float16* hb = (l == 0) ? h1 : h2;
        const _Float16* TT = (l == 0) ? T1 : T2;
        const _Float16* rT = (l == 0) ? r1 : r2;
        const float*  bias = (l == 0) ? b1 : b2;
        k_gather<<<(NE * 16 + 255) / 256, B, 0, stream>>>(xf, srce, xs);
        k_zero<<<(NN * DD + 255) / 256, B, 0, stream>>>(agg, NN * DD);
        k_gemm_msg<<<NE / 128, B, 0, stream>>>(xs, hb, TT, dste, agg);
        k_root_combine<<<NN / 64, B, 0, stream>>>(xf, rT, agg, cnt, bias, out,
                                                  (l < 2) ? 1 : 0, (l == 2) ? 1 : 0);
    }
}
